// TransGRUNet_1_4_85650237816875
// MI455X (gfx1250) — compile-verified, matrix operand order checked by reading, not run on hardware
//
#include <hip/hip_runtime.h>
#include <math.h>

// ---------------------------------------------------------------------------
// Types for CDNA5 WMMA (wave32): v_wmma_f32_16x16x32_bf16
// ---------------------------------------------------------------------------
typedef __attribute__((ext_vector_type(16))) __bf16 bf16x16;
typedef __attribute__((ext_vector_type(8)))  float  f32x8;
typedef __attribute__((ext_vector_type(4)))  unsigned int u32x4;
typedef __attribute__((ext_vector_type(8)))  int  i32x8;
typedef __attribute__((ext_vector_type(4)))  int  i32x4;

#define WMMA_BF16(a, b, c) \
  __builtin_amdgcn_wmma_f32_16x16x32_bf16(false, (a), false, (b), (short)0, (c), false, false)

#if defined(__has_builtin)
#if __has_builtin(__builtin_amdgcn_tensor_load_to_lds) && \
    __has_builtin(__builtin_amdgcn_s_wait_tensorcnt)
#define USE_TDM 1
#endif
#endif
#ifndef USE_TDM
#define USE_TDM 0
#endif

__device__ __forceinline__ float wave_sum32(float v) {
#pragma unroll
  for (int off = 16; off >= 1; off >>= 1) v += __shfl_xor(v, off, 32);
  return v;
}

__device__ __forceinline__ void atomicMaxF(float* addr, float val) {
  unsigned int* ua = (unsigned int*)addr;
  unsigned int old = *ua;
  while (true) {
    float cur = __uint_as_float(old);
    if (cur >= val) break;
    unsigned int assumed = old;
    old = atomicCAS(ua, assumed, __float_as_uint(val));
    if (old == assumed) break;
  }
}

__device__ __forceinline__ float sigmoidf(float x) { return 1.0f / (1.0f + expf(-x)); }

#if USE_TDM
// Issue one TDM DMA: copy a [tile_rows x tile_cols] bf16 tile (row stride K
// elements) from global memory into LDS at byte offset lds_addr.
// 2D D# per CDNA5 ISA 8.3/8.4.
__device__ __forceinline__ void tdm_load_tile_bf16(const __bf16* gptr, unsigned lds_addr,
                                                   int K, int tensor_rows,
                                                   int tile_cols, int tile_rows) {
  unsigned long long ga = (unsigned long long)(uintptr_t)gptr;
  u32x4 g0 = {
      1u,                                   // count=1, user desc
      lds_addr,                             // lds_addr (bytes)
      (unsigned)(ga & 0xFFFFFFFFu),         // global_addr[31:0]
      (unsigned)((ga >> 32) & 0x01FFFFFFu) | (2u << 30)  // addr[56:32] | type=2
  };
  unsigned uK = (unsigned)K, uM = (unsigned)tensor_rows;
  i32x8 g1 = {
      (int)(1u << 16),                                   // data_size=1 (2 bytes)
      (int)((uK & 0xFFFFu) << 16),                       // tensor_dim0[15:0]
      (int)(((uK >> 16) & 0xFFFFu) | ((uM & 0xFFFFu) << 16)),  // dim0 hi | dim1 lo
      (int)(((uM >> 16) & 0xFFFFu) | ((unsigned)tile_cols << 16)),  // dim1 hi | tile_dim0
      (int)tile_rows,                                    // tile_dim1 (tile_dim2=0)
      (int)uK,                                           // tensor_dim0_stride[31:0]
      0, 0
  };
  i32x4 gz = {0, 0, 0, 0};
#if __clang_major__ >= 23
  i32x8 gz8 = {0, 0, 0, 0, 0, 0, 0, 0};
  __builtin_amdgcn_tensor_load_to_lds(g0, g1, gz, gz, gz8, 0);
#else
  __builtin_amdgcn_tensor_load_to_lds(g0, g1, gz, gz, 0);
#endif
}
#endif

// ---------------------------------------------------------------------------
// Weight packing: fp32 [K][N] -> bf16 [N][K]  (B-fragment friendly layout)
// ---------------------------------------------------------------------------
__global__ void pack_t_k(const float* __restrict__ W, __bf16* __restrict__ P, int K, int N) {
  int t = blockIdx.x * blockDim.x + threadIdx.x;
  if (t >= K * N) return;
  int k = t / N, nn = t % N;
  P[(size_t)nn * K + k] = (__bf16)W[t];
}

// fp32 -> bf16, same layout
__global__ void pack_k(const float* __restrict__ W, __bf16* __restrict__ P, int n) {
  int t = blockIdx.x * blockDim.x + threadIdx.x;
  if (t >= n) return;
  P[t] = (__bf16)W[t];
}

// concat 4 bias vectors [d] -> [4d]
__global__ void concat4_k(float* __restrict__ dst, const float* __restrict__ b0,
                          const float* __restrict__ b1, const float* __restrict__ b2,
                          const float* __restrict__ b3, int d) {
  int t = blockIdx.x * blockDim.x + threadIdx.x;
  if (t >= 4 * d) return;
  int which = t / d, c = t - which * d;
  const float* b = (which == 0) ? b0 : (which == 1) ? b1 : (which == 2) ? b2 : b3;
  dst[t] = b[c];
}

// ---------------------------------------------------------------------------
// WMMA GEMM:  C[M,N] = A[M,K](bf16) * B[K,N] + bias,  B packed bf16 [N][K].
// Block: 256 threads = 8 waves (4x2 wave grid). Block tile 128x128,
// each wave 32x64 = 2x4 wmma 16x16 tiles, K-step 32.
// BOTH A and B tiles staged to LDS by the Tensor Data Mover (double
// buffered; wave0 -> A, wave1 -> B), overlapped with WMMA compute;
// sync via per-wave TENSORcnt + workgroup barrier. Inner loop is pure
// ds_load_b128 + v_wmma (no global VMEM).
// ---------------------------------------------------------------------------
#define TM 128
#define TN 128
#define TK 32

__global__ __launch_bounds__(256) void gemm_bias_wmma(
    const __bf16* __restrict__ A, const __bf16* __restrict__ Bp,
    const float* __restrict__ bias, float* __restrict__ C,
    int M, int K, int N) {
  __shared__ __bf16 sA[2][TM][TK];
  __shared__ __bf16 sB[2][TN][TK];

  const int tid  = threadIdx.x;
  const int lane = tid & 31;
  const int wave = tid >> 5;
  const int wm   = wave >> 1;  // 0..3  (32 rows each)
  const int wn   = wave & 1;   // 0..1  (64 cols each)
  const int m0   = blockIdx.x * TM;
  const int n0   = blockIdx.y * TN;

  const f32x8 zed = {0.f, 0.f, 0.f, 0.f, 0.f, 0.f, 0.f, 0.f};
  f32x8 acc[2][4];
#pragma unroll
  for (int i = 0; i < 2; i++)
#pragma unroll
    for (int j = 0; j < 4; j++) acc[i][j] = zed;

  const int arow = lane & 15;
  const int koff = (lane >> 4) << 4;
  const int T = K / TK;

#if USE_TDM
  const unsigned ldsA = (unsigned)(unsigned long long)(uintptr_t)&sA[0][0][0];
  const unsigned ldsB = (unsigned)(unsigned long long)(uintptr_t)&sB[0][0][0];
  const unsigned bufBytes = TM * TK * 2;
  if (wave == 0) tdm_load_tile_bf16(A + (size_t)m0 * K, ldsA, K, M, TK, TM);
  if (wave == 1) tdm_load_tile_bf16(Bp + (size_t)n0 * K, ldsB, K, N, TK, TN);
#endif

  for (int it = 0; it < T; it++) {
    const int cur = it & 1;
    const int kk = it * TK;

#if USE_TDM
    if (wave == 0) {
      if (it + 1 < T) {
        tdm_load_tile_bf16(A + (size_t)m0 * K + (kk + TK), ldsA + (cur ^ 1) * bufBytes,
                           K, M, TK, TM);
        __builtin_amdgcn_s_wait_tensorcnt(1);  // A tile `it` landed, next in flight
      } else {
        __builtin_amdgcn_s_wait_tensorcnt(0);
      }
    }
    if (wave == 1) {
      if (it + 1 < T) {
        tdm_load_tile_bf16(Bp + (size_t)n0 * K + (kk + TK), ldsB + (cur ^ 1) * bufBytes,
                           K, N, TK, TN);
        __builtin_amdgcn_s_wait_tensorcnt(1);  // B tile `it` landed, next in flight
      } else {
        __builtin_amdgcn_s_wait_tensorcnt(0);
      }
    }
    __syncthreads();
#else
    {
      const int ldRow = tid >> 1;        // 0..127
      const int ldCol = (tid & 1) * 16;  // 0 or 16
      int gr = m0 + ldRow;
      if (gr < M) {
        *(bf16x16*)&sA[cur][ldRow][ldCol] = *(const bf16x16*)(A + (size_t)gr * K + kk + ldCol);
      } else {
        const bf16x16 zb = {};
        *(bf16x16*)&sA[cur][ldRow][ldCol] = zb;
      }
      int gn = n0 + ldRow;
      if (gn < N) {
        *(bf16x16*)&sB[cur][ldRow][ldCol] = *(const bf16x16*)(Bp + (size_t)gn * K + kk + ldCol);
      } else {
        const bf16x16 zb = {};
        *(bf16x16*)&sB[cur][ldRow][ldCol] = zb;
      }
    }
    __syncthreads();
#endif

    bf16x16 bfrag[4];
#pragma unroll
    for (int j = 0; j < 4; j++)
      bfrag[j] = *(const bf16x16*)&sB[cur][wn * 64 + j * 16 + arow][koff];
#pragma unroll
    for (int i = 0; i < 2; i++) {
      bf16x16 afrag = *(const bf16x16*)&sA[cur][wm * 32 + i * 16 + arow][koff];
#pragma unroll
      for (int j = 0; j < 4; j++) acc[i][j] = WMMA_BF16(afrag, bfrag[j], acc[i][j]);
    }
    __syncthreads();  // readers done before these buffers are overwritten
  }

  // C/D layout: VGPR r -> M = r (lanes 0-15) / r+8 (lanes 16-31), N = lane&15
  const int rsel = (lane >> 4) << 3;
#pragma unroll
  for (int i = 0; i < 2; i++) {
#pragma unroll
    for (int j = 0; j < 4; j++) {
      int col = n0 + wn * 64 + j * 16 + (lane & 15);
      float bv = bias ? bias[col] : 0.f;
#pragma unroll
      for (int r = 0; r < 8; r++) {
        int row = m0 + wm * 32 + i * 16 + r + rsel;
        if (row < M) C[(size_t)row * N + col] = acc[i][j][r] + bv;
      }
    }
  }
}

// ---------------------------------------------------------------------------
// TransformerConv attention kernels. q/k/v/skip live in one fused buffer
// qkvs[n][ld] with ld = 4*dout; column slices: q=0, k=dout, v=2*dout,
// skip/agg=3*dout (skip output doubles as attention accumulator).
// ---------------------------------------------------------------------------
__global__ void fill_f(float* __restrict__ p, float v, int n) {
  int i = blockIdx.x * blockDim.x + threadIdx.x;
  if (i < n) p[i] = v;
}

__global__ void edge_logit_k(const float* __restrict__ qkvs,
                             const int* __restrict__ src, const int* __restrict__ dst,
                             float* __restrict__ logit, int E, int ld, int dout) {
  int e = (blockIdx.x * blockDim.x + threadIdx.x) >> 5;
  int lane = threadIdx.x & 31;
  if (e >= E) return;
  const float* qr = qkvs + (size_t)dst[e] * ld;
  const float* kr = qkvs + (size_t)src[e] * ld + dout;
  float s = 0.f;
  for (int i = lane; i < dout; i += 32) s += qr[i] * kr[i];
  s = wave_sum32(s);
  if (lane == 0) logit[e] = s * rsqrtf((float)dout);
}

__global__ void seg_max_k(const float* __restrict__ logit, const int* __restrict__ dst,
                          float* __restrict__ m, int E) {
  int e = blockIdx.x * blockDim.x + threadIdx.x;
  if (e >= E) return;
  atomicMaxF(&m[dst[e]], logit[e]);
}

__global__ void edge_exp_k(const float* __restrict__ logit, const int* __restrict__ dst,
                           const float* __restrict__ m, float* __restrict__ ebuf,
                           float* __restrict__ denom, int E) {
  int e = blockIdx.x * blockDim.x + threadIdx.x;
  if (e >= E) return;
  float v = expf(logit[e] - m[dst[e]]);
  ebuf[e] = v;
  atomicAdd(&denom[dst[e]], v);
}

__global__ void edge_agg_k(const float* __restrict__ ebuf, const float* __restrict__ denom,
                           float* qkvs, const int* __restrict__ src,
                           const int* __restrict__ dst, int E, int ld, int dout) {
  int t = blockIdx.x * blockDim.x + threadIdx.x;
  int perEdge = dout >> 2;
  int e = t / perEdge;
  if (e >= E) return;
  int f0 = (t - e * perEdge) * 4;
  float attn = ebuf[e] / denom[dst[e]];
  const float* vr = qkvs + (size_t)src[e] * ld + 2 * dout + f0;
  float* ar = qkvs + (size_t)dst[e] * ld + 3 * dout + f0;
#pragma unroll
  for (int j = 0; j < 4; j++) atomicAdd(&ar[j], attn * vr[j]);
}

// LayerNorm (+optional relu), one wave per row; strided input, optional bf16 twin
__global__ void ln_act(const float* __restrict__ X, const float* __restrict__ gw,
                       const float* __restrict__ bw, float* __restrict__ Y,
                       __bf16* __restrict__ Ybf, int rows, int d, int relu, int ldx) {
  int row = (blockIdx.x * blockDim.x + threadIdx.x) >> 5;
  int lane = threadIdx.x & 31;
  if (row >= rows) return;
  const float* xr = X + (size_t)row * ldx;
  float s = 0.f, s2 = 0.f;
  for (int i = lane; i < d; i += 32) { float v = xr[i]; s += v; s2 += v * v; }
  s = wave_sum32(s);
  s2 = wave_sum32(s2);
  float mean = s / (float)d;
  float var = s2 / (float)d - mean * mean;
  float inv = rsqrtf(var + 1e-5f);
  float* yr = Y + (size_t)row * d;
  for (int i = lane; i < d; i += 32) {
    float v = (xr[i] - mean) * inv * gw[i] + bw[i];
    if (relu) v = fmaxf(v, 0.f);
    yr[i] = v;
    if (Ybf) Ybf[(size_t)row * d + i] = (__bf16)v;
  }
}

// ---------------------------------------------------------------------------
// Cluster assignment (cosine sim to 4 centers) + pairwise-dist variance
// ---------------------------------------------------------------------------
__global__ void norm_centers_k(const float* __restrict__ centers, float* __restrict__ cn) {
  int c = threadIdx.x >> 5;  // 4 waves, block 128
  int lane = threadIdx.x & 31;
  float s = 0.f;
  for (int i = lane; i < 256; i += 32) { float v = centers[c * 256 + i]; s += v * v; }
  s = wave_sum32(s);
  float inv = 1.0f / fmaxf(sqrtf(s), 1e-8f);
  for (int i = lane; i < 256; i += 32) cn[c * 256 + i] = centers[c * 256 + i] * inv;
}

__global__ void assign_k(const float* __restrict__ h, const float* __restrict__ cn,
                         int* __restrict__ level, int n) {
  int node = (blockIdx.x * blockDim.x + threadIdx.x) >> 5;
  int lane = threadIdx.x & 31;
  if (node >= n) return;
  const float* hr = h + (size_t)node * 256;
  float s = 0.f;
  for (int i = lane; i < 256; i += 32) { float v = hr[i]; s += v * v; }
  s = wave_sum32(s);
  float inv = 1.0f / fmaxf(sqrtf(s), 1e-8f);
  float best = -1e30f;
  int bi = 0;
#pragma unroll
  for (int c = 0; c < 4; c++) {
    float dsum = 0.f;
    for (int i = lane; i < 256; i += 32) dsum += hr[i] * cn[c * 256 + i];
    dsum = wave_sum32(dsum);
    float sim = dsum * inv;
    if (sim > best) { best = sim; bi = c; }
  }
  if (lane == 0) level[node] = bi;
}

__global__ void distance_k(const float* __restrict__ centers, float* __restrict__ out) {
  float dists[6];
  int idx = 0;
  for (int i = 0; i < 4; i++)
    for (int j = i + 1; j < 4; j++) {
      float s = 0.f;
      for (int k = 0; k < 256; k++) {
        float dd = centers[i * 256 + k] - centers[j * 256 + k];
        s += dd * dd;
      }
      dists[idx++] = sqrtf(s);
    }
  float mean = 0.f;
  for (int i = 0; i < 6; i++) mean += dists[i];
  mean /= 6.f;
  float var = 0.f;
  for (int i = 0; i < 6; i++) { float d = dists[i] - mean; var += d * d; }
  var /= 5.f;  // ddof=1
  out[0] = -var;
}

// ---------------------------------------------------------------------------
// Ragged (graph, cluster) grouping
// ---------------------------------------------------------------------------
__global__ void group_k(const int* __restrict__ batch, const int* __restrict__ level,
                        int* __restrict__ g, int* __restrict__ counts, int n) {
  int i = blockIdx.x * blockDim.x + threadIdx.x;
  if (i >= n) return;
  int gg = batch[i] * 4 + level[i];
  g[i] = gg;
  atomicAdd(&counts[gg], 1);
}

// exact lexsort rank within group: key = (call_seq, node index)
__global__ void rank_k(const int* __restrict__ g, const int* __restrict__ cs,
                       int* __restrict__ pos, int n) {
  int i = blockIdx.x * blockDim.x + threadIdx.x;
  if (i >= n) return;
  int gi = g[i], ci = cs[i], p = 0;
  for (int j = 0; j < n; j++) {
    if (g[j] == gi && (cs[j] < ci || (cs[j] == ci && j < i))) p++;
  }
  pos[i] = p;
}

__global__ void scatter_k(const float* __restrict__ h, const int* __restrict__ g,
                          const int* __restrict__ pos, __bf16* __restrict__ feats_bf,
                          int n, int S) {
  long long t = (long long)blockIdx.x * blockDim.x + threadIdx.x;
  if (t >= (long long)n * 256) return;
  int i = (int)(t >> 8);
  int f = (int)(t & 255);
  int p = pos[i];
  if (p >= S) return;
  feats_bf[((size_t)g[i] * S + p) * 256 + f] = (__bf16)h[(size_t)i * 256 + f];
}

__global__ void add_pe_k(const float* __restrict__ proj, __bf16* __restrict__ proj_bf,
                         long long total, int S) {
  long long t = (long long)blockIdx.x * blockDim.x + threadIdx.x;
  if (t >= total) return;
  int c = (int)(t & 255);
  long long rs = t >> 8;
  int s = (int)(rs % S);
  int i2 = c & ~1;
  float div = expf(-(float)i2 * (logf(10000.f) / 256.f));
  float a = (float)s * div;
  float pe = (c & 1) ? cosf(a) : sinf(a);
  proj_bf[t] = (__bf16)(proj[t] + pe);
}

// ---------------------------------------------------------------------------
// GRU scan with WMMA: each block owns 16 groups; per time-step computes
// gh = h @ W_hh^T  ([16,256] x [256,768]) via v_wmma_f32_16x16x32_bf16.
// ---------------------------------------------------------------------------
__global__ __launch_bounds__(256) void gru_scan_wmma(
    const float* __restrict__ xi, const __bf16* __restrict__ whh,
    const float* __restrict__ bhh, float* __restrict__ hs, int S) {
  __shared__ float  h_f[16][256];
  __shared__ __bf16 h_b[16][256];
  __shared__ float  gh[16][768];

  const int tid  = threadIdx.x;
  const int lane = tid & 31;
  const int wave = tid >> 5;
  const int g0   = blockIdx.x * 16;
  const int arow = lane & 15;
  const int koff = (lane >> 4) << 4;
  const int rsel = (lane >> 4) << 3;
  const f32x8 zed = {0.f, 0.f, 0.f, 0.f, 0.f, 0.f, 0.f, 0.f};

  for (int i = tid; i < 16 * 256; i += 256) (&h_f[0][0])[i] = 0.f;
  __syncthreads();

  for (int t = 0; t < S; t++) {
    for (int i = tid; i < 16 * 256; i += 256)
      (&h_b[0][0])[i] = (__bf16)(&h_f[0][0])[i];
    __syncthreads();

    for (int tno = 0; tno < 6; tno++) {
      int nb = (wave * 6 + tno) * 16;
      f32x8 acc = zed;
#pragma unroll
      for (int k0 = 0; k0 < 256; k0 += 32) {
        bf16x16 afrag = *(const bf16x16*)&h_b[arow][k0 + koff];
        bf16x16 bfrag = *(const bf16x16*)(whh + (size_t)(nb + arow) * 256 + k0 + koff);
        acc = WMMA_BF16(afrag, bfrag, acc);
      }
      int col = nb + (lane & 15);
#pragma unroll
      for (int r = 0; r < 8; r++) gh[r + rsel][col] = acc[r];
    }
    __syncthreads();

    for (int i = tid; i < 16 * 256; i += 256) {
      int g = i >> 8, k = i & 255;
      const float* xrow = xi + ((size_t)(g0 + g) * S + t) * 768;
      float xr = xrow[k], xz = xrow[256 + k], xn = xrow[512 + k];
      float hr = gh[g][k] + bhh[k];
      float hz = gh[g][256 + k] + bhh[256 + k];
      float hn = gh[g][512 + k] + bhh[512 + k];
      float r = sigmoidf(xr + hr);
      float z = sigmoidf(xz + hz);
      float nnv = tanhf(xn + r * hn);
      float hp = h_f[g][k];
      float hnew = (1.f - z) * nnv + z * hp;
      h_f[g][k] = hnew;
      hs[((size_t)(g0 + g) * S + t) * 256 + k] = hnew;
    }
    __syncthreads();
  }
}

// ---------------------------------------------------------------------------
// Final: select rnn/raw features, 256->8 gelu -> 8->1 sigmoid
// ---------------------------------------------------------------------------
__global__ void final_k(const float* __restrict__ h, const float* __restrict__ rnn,
                        const int* __restrict__ g, const int* __restrict__ pos,
                        const int* __restrict__ counts,
                        const float* __restrict__ w1, const float* __restrict__ b1,
                        const float* __restrict__ w2, const float* __restrict__ b2,
                        float* __restrict__ out, int n, int S) {
  int i = blockIdx.x * blockDim.x + threadIdx.x;
  if (i >= n) return;
  int gg = g[i], p = pos[i];
  const float* f = (counts[gg] > 1 && p < S) ? rnn + ((size_t)gg * S + p) * 256
                                             : h + (size_t)i * 256;
  float t8[8];
#pragma unroll
  for (int j = 0; j < 8; j++) t8[j] = b1[j];
  for (int k = 0; k < 256; k++) {
    float v = f[k];
#pragma unroll
    for (int j = 0; j < 8; j++) t8[j] += v * w1[k * 8 + j];
  }
  float o = b2[0];
#pragma unroll
  for (int j = 0; j < 8; j++) {
    float gel = 0.5f * t8[j] * (1.f + erff(t8[j] * 0.70710678118654752f));
    o += gel * w2[j];
  }
  out[i] = sigmoidf(o);
}

// ---------------------------------------------------------------------------
// Host orchestration
// ---------------------------------------------------------------------------
static inline unsigned int cdivu(long long a, long long b) {
  return (unsigned int)((a + b - 1) / b);
}

extern "C" void kernel_launch(void* const* d_in, const int* in_sizes, int n_in,
                              void* d_out, int out_size, void* d_ws, size_t ws_size,
                              hipStream_t stream) {
  (void)n_in; (void)out_size; (void)ws_size;

  const int D0 = 256;
  const int n = in_sizes[0] / D0;  // 32768
  const float* x = (const float*)d_in[0];
  auto P = [&](int i) { return (const float*)d_in[i]; };

  const int* edge = (const int*)d_in[54];
  const int E = in_sizes[54] / 2;
  const int* srcI = edge;
  const int* dstI = edge + E;
  const int* csI = (const int*)d_in[55];
  const int* batchI = (const int*)d_in[56];

  const int S = 512, G = 256;  // B=64, L=4
  const long long GS = (long long)G * S;

  float* outp = (float*)d_out;

  size_t off = 0;
  auto alloc = [&](size_t bytes) -> char* {
    char* p = (char*)d_ws + off;
    off += (bytes + 255) & ~(size_t)255;
    return p;
  };

  float* hA     = (float*)alloc((size_t)n * 512 * 4);
  float* hB     = (float*)alloc((size_t)n * 512 * 4);
  __bf16* x_bf  = (__bf16*)alloc((size_t)n * 256 * 2);
  __bf16* hA_bf = (__bf16*)alloc((size_t)n * 512 * 2);
  __bf16* hB_bf = (__bf16*)alloc((size_t)n * 512 * 2);
  float* qkvs   = (float*)alloc((size_t)n * 2048 * 4);   // fused q|k|v|skip
  float* bias4  = (float*)alloc((size_t)2048 * 4);
  float* logit  = (float*)alloc((size_t)E * 4);
  float* ebuf   = (float*)alloc((size_t)E * 4);
  float* mbuf   = (float*)alloc((size_t)n * 4);
  float* denom  = (float*)alloc((size_t)n * 4);
  __bf16* wpack = (__bf16*)alloc((size_t)512 * 512 * 4 * 2);  // [4*dout][din] bf16
  float* cn     = (float*)alloc(4 * 256 * 4);
  int* level    = (int*)alloc((size_t)n * 4);
  int* gidx     = (int*)alloc((size_t)n * 4);
  int* pos      = (int*)alloc((size_t)n * 4);
  int* counts   = (int*)alloc((size_t)G * 4);
  __bf16* feats_bf = (__bf16*)alloc((size_t)GS * 256 * 2);
  float* proj   = (float*)alloc((size_t)GS * 256 * 4);
  __bf16* proj_bf = (__bf16*)alloc((size_t)GS * 256 * 2);
  float* xib    = (float*)alloc((size_t)GS * 768 * 4);
  float* hsb    = (float*)alloc((size_t)GS * 256 * 4);
  float* rnnb   = (float*)alloc((size_t)GS * 256 * 4);
  __bf16* wip   = (__bf16*)alloc((size_t)256 * 256 * 2);
  __bf16* wihp  = (__bf16*)alloc((size_t)768 * 256 * 2);
  __bf16* whhp  = (__bf16*)alloc((size_t)768 * 256 * 2);

  // bf16 twin of the network input
  pack_k<<<cdivu((long long)n * 256, 256), 256, 0, stream>>>(x, x_bf, n * 256);

  // ---- 4 TransformerConv layers (fused q|k|v|skip GEMM per layer) ----
  const float* hin = x;
  const __bf16* hin_bf = x_bf;
  const int dims[5] = {256, 512, 512, 512, 256};
  for (int layer = 0; layer < 4; layer++) {
    int din = dims[layer], dout = dims[layer + 1];
    int N4 = 4 * dout;
    int base = 1 + layer * 10;
    const float *qw = P(base),     *qbi = P(base + 1),
                *kw = P(base + 2), *kbi = P(base + 3),
                *vw = P(base + 4), *vbi = P(base + 5),
                *sw = P(base + 6), *sbi = P(base + 7),
                *lng = P(base + 8), *lnb = P(base + 9);
    size_t kn = (size_t)din * dout;
    // pack into one B = [q rows | k rows | v rows | skip rows][din]
    pack_t_k<<<cdivu(kn, 256), 256, 0, stream>>>(qw, wpack + 0 * kn, din, dout);
    pack_t_k<<<cdivu(kn, 256), 256, 0, stream>>>(kw, wpack + 1 * kn, din, dout);
    pack_t_k<<<cdivu(kn, 256), 256, 0, stream>>>(vw, wpack + 2 * kn, din, dout);
    pack_t_k<<<cdivu(kn, 256), 256, 0, stream>>>(sw, wpack + 3 * kn, din, dout);
    concat4_k<<<cdivu(N4, 256), 256, 0, stream>>>(bias4, qbi, kbi, vbi, sbi, dout);

    dim3 gg(cdivu(n, TM), cdivu(N4, TN));
    gemm_bias_wmma<<<gg, 256, 0, stream>>>(hin_bf, wpack, bias4, qkvs, n, din, N4);

    edge_logit_k<<<cdivu((long long)E * 32, 256), 256, 0, stream>>>(qkvs, srcI, dstI, logit, E, N4, dout);
    fill_f<<<cdivu(n, 256), 256, 0, stream>>>(mbuf, -3.0e38f, n);
    hipMemsetAsync(denom, 0, (size_t)n * 4, stream);
    seg_max_k<<<cdivu(E, 256), 256, 0, stream>>>(logit, dstI, mbuf, E);
    edge_exp_k<<<cdivu(E, 256), 256, 0, stream>>>(logit, dstI, mbuf, ebuf, denom, E);
    edge_agg_k<<<cdivu((long long)E * (dout / 4), 256), 256, 0, stream>>>(
        ebuf, denom, qkvs, srcI, dstI, E, N4, dout);

    float* hout = (layer & 1) ? hB : hA;
    __bf16* hout_bf = (layer & 1) ? hB_bf : hA_bf;
    ln_act<<<cdivu((long long)n * 32, 256), 256, 0, stream>>>(
        qkvs + 3 * dout, lng, lnb, hout, hout_bf, n, dout, 1, N4);
    hin = hout;
    hin_bf = hout_bf;
  }

  // ---- cluster assignment + distance ----
  norm_centers_k<<<1, 128, 0, stream>>>(P(41), cn);
  assign_k<<<cdivu((long long)n * 32, 256), 256, 0, stream>>>(hin, cn, level, n);
  distance_k<<<1, 1, 0, stream>>>(P(41), outp + n);

  // ---- grouping ----
  hipMemsetAsync(counts, 0, (size_t)G * 4, stream);
  group_k<<<cdivu(n, 256), 256, 0, stream>>>(batchI, level, gidx, counts, n);
  rank_k<<<cdivu(n, 256), 256, 0, stream>>>(gidx, csI, pos, n);
  hipMemsetAsync(feats_bf, 0, (size_t)GS * 256 * 2, stream);
  scatter_k<<<cdivu((long long)n * 256, 256), 256, 0, stream>>>(hin, gidx, pos, feats_bf, n, S);

  // ---- sequential RNN path ----
  pack_t_k<<<cdivu(256 * 256, 256), 256, 0, stream>>>(P(42), wip, 256, 256);
  gemm_bias_wmma<<<dim3(cdivu(GS, TM), cdivu(256, TN)), 256, 0, stream>>>(
      feats_bf, wip, P(43), proj, (int)GS, 256, 256);
  add_pe_k<<<cdivu(GS * 256, 256), 256, 0, stream>>>(proj, proj_bf, GS * 256, S);

  pack_k<<<cdivu(768 * 256, 256), 256, 0, stream>>>(P(44), wihp, 768 * 256);  // w_ih [768][256]
  gemm_bias_wmma<<<dim3(cdivu(GS, TM), cdivu(768, TN)), 256, 0, stream>>>(
      proj_bf, wihp, P(46), xib, (int)GS, 256, 768);

  pack_k<<<cdivu(768 * 256, 256), 256, 0, stream>>>(P(45), whhp, 768 * 256);  // w_hh [768][256]
  gru_scan_wmma<<<G / 16, 256, 0, stream>>>(xib, whhp, P(47), hsb, S);

  ln_act<<<cdivu(GS * 32, 256), 256, 0, stream>>>(hsb, P(48), P(49), rnnb, nullptr, (int)GS, 256, 1, 256);

  final_k<<<cdivu(n, 256), 256, 0, stream>>>(hin, rnnb, gidx, pos, counts,
                                             P(50), P(51), P(52), P(53), outp, n, S);
}
